// Seq2Seq_55448027791456
// MI455X (gfx1250) — compile-verified, matrix operand order checked by reading, not run on hardware
//
#include <hip/hip_runtime.h>

// ---------------------------------------------------------------------------
// Seq2Seq LSTM (L=3, H=256, B=64, T=64, V=32000) for MI455X (gfx1250).
//  - bf16 WMMA (v_wmma_f32_16x16x32_bf16) for every GEMM, f32 accumulate.
//  - GEMM: 2x4 tiles/wave register blocking + block-shared B panel staged in
//    LDS (32KB). Per block (256x64 C patch): A 128KB from L2, B 32KB once ->
//    FC L2 read demand ~58 TB/s inside its 22us HBM-write window
//    (524MB / 23.3 TB/s), instead of ~380 TB/s for the naive 1-tile/wave form.
//  - Input projections x@Wih^T batched over all T (bias folded in).
//  - Per-layer recurrence in ONE persistent 32-wave workgroup: each wave owns
//    one m-tile x two j-tiles and ALL FOUR gate tiles for them, so the LSTM
//    pointwise update runs straight out of WMMA accumulators; c in LDS (f32),
//    h double-buffered in LDS (bf16), one barrier per timestep.
//  - wave32-shuffle argmax appended after logits.
// ---------------------------------------------------------------------------

typedef __attribute__((ext_vector_type(16))) __bf16 v16bf;
typedef __attribute__((ext_vector_type(8)))  __bf16 v8bf;
typedef __attribute__((ext_vector_type(8)))  float  v8f;

#define LL   3
#define HH   256
#define BB   64
#define TT   64
#define G4H  1024     // 4*H
#define VV   32000

__device__ __forceinline__ unsigned short f2bf(float f) {
    unsigned int u = __builtin_bit_cast(unsigned int, f);
    unsigned int r = u + 0x7FFFu + ((u >> 16) & 1u);   // round-to-nearest-even
    return (unsigned short)(r >> 16);
}

__device__ __forceinline__ v16bf mk16(v8bf lo, v8bf hi) {
    return __builtin_shufflevector(lo, hi, 0,1,2,3,4,5,6,7,8,9,10,11,12,13,14,15);
}

__device__ __forceinline__ v16bf ldfrag(const unsigned short* p) {
    return mk16(*(const v8bf*)p, *(const v8bf*)(p + 16));
}

__device__ __forceinline__ float sigf(float x) {
    return 1.0f / (1.0f + __expf(-x));
}

// ---------------- generic f32 -> bf16 convert ------------------------------
__global__ __launch_bounds__(256)
void k_f32_to_bf16(const float* __restrict__ src, unsigned short* __restrict__ dst, int n) {
    int i = blockIdx.x * blockDim.x + threadIdx.x;
    if (i < n) dst[i] = f2bf(src[i]);
}

// ---------------- embedding gather (optionally shifted for decoder) --------
__global__ __launch_bounds__(256)
void k_embed(const int* __restrict__ tokens, const float* __restrict__ emb,
             unsigned short* __restrict__ X, int shift) {
    int i = blockIdx.x * blockDim.x + threadIdx.x;   // T*B*H
    if (i >= TT * BB * HH) return;
    int j  = i % HH;
    int tb = i / HH;
    int b  = tb % BB;
    int t  = tb / BB;
    int tok;
    if (shift) tok = (t == 0) ? 0 : tokens[(t - 1) * BB + b];
    else       tok = tokens[t * BB + b];
    X[i] = f2bf(emb[(long)tok * HH + j]);
}

// ---------------- broadcast learned initial states -------------------------
__global__ __launch_bounds__(256)
void k_init_state(const float* __restrict__ h0, const float* __restrict__ c0,
                  float* __restrict__ h, float* __restrict__ c) {
    int i = blockIdx.x * blockDim.x + threadIdx.x;   // L*B*H
    if (i >= LL * BB * HH) return;
    int j = i % HH;
    int l = i / (BB * HH);
    h[i] = h0[l * HH + j];
    c[i] = c0[l * HH + j];
}

// ---------------- WMMA bf16 GEMM, LDS-shared B panel ------------------------
// C[M,N] = A[M,K=256] * Bw[N,256]^T (+ bias0[n] + bias1[n]).
// Block = 256 threads = 8 waves covering a 256x64 C patch: one shared 64x256
// bf16 B panel staged in LDS (32KB), each wave computes 2 m-tiles x 4 n-tiles
// (A fragments from global/L2, B fragments from LDS; 8 WMMAs per k-step).
// Requires M%256==0, N%64==0; grids are exact so there are no divergent
// guards and EXEC is all-ones at every WMMA.
__global__ __launch_bounds__(256)
void k_gemm_blk(const unsigned short* __restrict__ A,   // [M,256] row-major bf16
                const unsigned short* __restrict__ Bw,  // [N,256] row-major bf16
                float* __restrict__ C,                  // [M,N]
                int M, int N,
                const float* __restrict__ bias0,        // [N] or nullptr
                const float* __restrict__ bias1)        // [N] or nullptr
{
    __shared__ unsigned short ldsB[64 * HH];            // 32KB B panel

    const int nbn = N >> 6;                   // 64-wide n-blocks
    const int bnb = blockIdx.x % nbn;
    const int bmb = blockIdx.x / nbn;         // 256-wide m-super-block
    const int tid = threadIdx.x;

    // stage B panel: rows [bnb*64, bnb*64+64), K=256  (vectorized 16B copies)
    {
        const v8bf* src = (const v8bf*)(Bw + (size_t)bnb * 64 * HH);
        v8bf*       dst = (v8bf*)ldsB;
        for (int idx = tid; idx < 64 * HH / 8; idx += 256) dst[idx] = src[idx];
    }
    __syncthreads();

    const int wv   = tid >> 5;
    const int lane = tid & 31;
    const int half = lane >> 4;               // 0: lanes 0-15, 1: lanes 16-31
    const int r    = lane & 15;
    const int bm   = bmb * 8 + wv;            // this wave's 32-wide m-block

    const unsigned short* a0 = A + (size_t)(bm * 32 + r) * HH;   // m-tile 0
    const unsigned short* a1 = a0 + (size_t)16 * HH;             // m-tile 1

    v8f acc[2][4];
#pragma unroll
    for (int mi = 0; mi < 2; ++mi)
#pragma unroll
        for (int j = 0; j < 4; ++j) acc[mi][j] = (v8f){};

    for (int k0 = 0; k0 < HH; k0 += 32) {
        const int ka = k0 + half * 8;
        v16bf af0 = ldfrag(a0 + ka);                              // global/L2
        v16bf af1 = ldfrag(a1 + ka);
#pragma unroll
        for (int j = 0; j < 4; ++j) {
            v16bf bf_ = ldfrag(ldsB + (j * 16 + r) * HH + ka);    // LDS
            acc[0][j] = __builtin_amdgcn_wmma_f32_16x16x32_bf16(false, af0, false, bf_,
                                                                (short)0, acc[0][j], false, false);
            acc[1][j] = __builtin_amdgcn_wmma_f32_16x16x32_bf16(false, af1, false, bf_,
                                                                (short)0, acc[1][j], false, false);
        }
    }

#pragma unroll
    for (int j = 0; j < 4; ++j) {
        const int n = bnb * 64 + j * 16 + r;
        float badd = 0.0f;
        if (bias0) badd += bias0[n];
        if (bias1) badd += bias1[n];
#pragma unroll
        for (int mi = 0; mi < 2; ++mi) {
#pragma unroll
            for (int i = 0; i < 8; ++i) {
                int m = bm * 32 + mi * 16 + half * 8 + i;  // C/D: VGPR i -> M=i(+8)
                C[(size_t)m * N + n] = acc[mi][j][i] + badd;
            }
        }
    }
}

// ---------------- persistent per-layer LSTM recurrence ----------------------
// One workgroup, 1024 threads = 32 wave32s. Time loop inside, one barrier per
// step. Wave w owns m-tile tm=w>>3 and j-tiles {2*(w&7), 2*(w&7)+1}; for that
// 16x32 patch it computes ALL FOUR gate tiles (columns g*256+j), so the LSTM
// pointwise update needs no gate staging. One LDS A-fragment feeds 8 WMMAs
// per k-step. c: LDS f32. h: double-buffered LDS bf16.
// h_state holds h_in on entry; final h/c written back at t==T-1.
__global__ __launch_bounds__(1024)
void k_lstm_layer(const unsigned short* __restrict__ Whh,  // [4H,H] bf16
                  const float* __restrict__ precomp,       // [T,B,4H], bias folded
                  float* __restrict__ h_state,             // [B,H] f32 in/out
                  float* __restrict__ c_state,             // [B,H] f32 in/out
                  unsigned short* __restrict__ Y)          // [T,B,H] bf16 out
{
    extern __shared__ char smem[];
    float*          cL  = (float*)smem;                          // 64KB
    unsigned short* hb0 = (unsigned short*)(smem + BB * HH * 4); // 32KB
    unsigned short* hb1 = hb0 + BB * HH;                         // 32KB

    const int tid  = threadIdx.x;
    const int wv   = tid >> 5;
    const int lane = tid & 31;
    const int half = lane >> 4;
    const int r    = lane & 15;

    const int tm  = wv >> 3;          // 0..3   (batch tile)
    const int tjb = (wv & 7) * 2;     // 0,2,..14 (first hidden-col tile)

    for (int i = tid; i < BB * HH; i += 1024) {
        cL[i]  = c_state[i];
        hb0[i] = f2bf(h_state[i]);
    }
    __syncthreads();

    unsigned short* hcur = hb0;
    unsigned short* hnxt = hb1;

    for (int t = 0; t < TT; ++t) {
        const float*    P  = precomp + (size_t)t * BB * G4H;
        unsigned short* Yt = Y + (size_t)t * BB * HH;
        const bool last = (t == TT - 1);

        const unsigned short* arow = hcur + (tm * 16 + r) * HH;
        v8f acc[2][4];
#pragma unroll
        for (int jj = 0; jj < 2; ++jj)
#pragma unroll
            for (int g = 0; g < 4; ++g) acc[jj][g] = (v8f){};

        for (int k0 = 0; k0 < HH; k0 += 32) {
            const int ka = k0 + half * 8;
            v16bf a = ldfrag(arow + ka);                     // LDS (ds_load_b128)
#pragma unroll
            for (int jj = 0; jj < 2; ++jj) {
                const unsigned short* brow =
                    Whh + (size_t)((tjb + jj) * 16 + r) * HH + ka;  // gate 0 row
#pragma unroll
                for (int g = 0; g < 4; ++g) {
                    v16bf b = ldfrag(brow + (size_t)g * HH * HH);
                    acc[jj][g] = __builtin_amdgcn_wmma_f32_16x16x32_bf16(
                        false, a, false, b, (short)0, acc[jj][g], false, false);
                }
            }
        }

        // pointwise LSTM update straight from accumulators (owned patch)
#pragma unroll
        for (int jj = 0; jj < 2; ++jj) {
            const int n = (tjb + jj) * 16 + r;     // hidden index j
#pragma unroll
            for (int i = 0; i < 8; ++i) {
                const int m = tm * 16 + half * 8 + i;   // batch index b
                const float* prow = P + (size_t)m * G4H;
                float gi = acc[jj][0][i] + prow[0 * HH + n];
                float gf = acc[jj][1][i] + prow[1 * HH + n];
                float gg = acc[jj][2][i] + prow[2 * HH + n];
                float go = acc[jj][3][i] + prow[3 * HH + n];
                float tg = tanhf(gg);
                const int ci = m * HH + n;
                float cn = sigf(gf) * cL[ci] + sigf(gi) * tg;
                float hn = sigf(go) * tanhf(cn);
                cL[ci] = cn;
                unsigned short hb = f2bf(hn);
                hnxt[ci] = hb;
                Yt[ci]   = hb;
                if (last) { h_state[ci] = hn; c_state[ci] = cn; }
            }
        }
        __syncthreads();   // hnxt complete & hcur reads done before swap
        unsigned short* sw = hcur; hcur = hnxt; hnxt = sw;
    }
}

// ---------------- row argmax over V (one wave32 per row) --------------------
__global__ __launch_bounds__(256)
void k_argmax(const float* __restrict__ logits, float* __restrict__ out) {
    int row = blockIdx.x * (blockDim.x >> 5) + (threadIdx.x >> 5);
    if (row >= TT * BB) return;
    int lane = threadIdx.x & 31;
    const float* p = logits + (long)row * VV;
    float bv = -3.402823466e38f;
    int   bi = 0;
    for (int n = lane; n < VV; n += 32) {
        float v = p[n];
        if (v > bv) { bv = v; bi = n; }
    }
    for (int off = 16; off > 0; off >>= 1) {
        float ov = __shfl_xor(bv, off, 32);
        int   oi = __shfl_xor(bi, off, 32);
        if (ov > bv || (ov == bv && oi < bi)) { bv = ov; bi = oi; }
    }
    if (lane == 0) out[row] = (float)bi;
}

// ---------------------------------------------------------------------------
extern "C" void kernel_launch(void* const* d_in, const int* in_sizes, int n_in,
                              void* d_out, int out_size, void* d_ws, size_t ws_size,
                              hipStream_t stream) {
    (void)in_sizes; (void)n_in; (void)out_size; (void)ws_size;

    const int*   inputs  = (const int*)  d_in[0];
    const int*   targets = (const int*)  d_in[1];
    const float* enc_emb = (const float*)d_in[2];
    const float* enc_Wih = (const float*)d_in[3];
    const float* enc_Whh = (const float*)d_in[4];
    const float* enc_bih = (const float*)d_in[5];
    const float* enc_bhh = (const float*)d_in[6];
    const float* hidden0 = (const float*)d_in[7];
    const float* cell0   = (const float*)d_in[8];
    const float* dec_emb = (const float*)d_in[9];
    const float* dec_Wih = (const float*)d_in[10];
    const float* dec_Whh = (const float*)d_in[11];
    const float* dec_bih = (const float*)d_in[12];
    const float* dec_bhh = (const float*)d_in[13];
    const float* fc_W    = (const float*)d_in[14];
    const float* fc_b    = (const float*)d_in[15];

    // ---- workspace carve-up (all offsets 256B aligned) ----
    char* ws = (char*)d_ws;
    size_t o = 0;
    const size_t nLW = (size_t)LL * G4H * HH;         // one weight stack
    unsigned short* wihE = (unsigned short*)(ws + o); o += nLW * 2;
    unsigned short* whhE = (unsigned short*)(ws + o); o += nLW * 2;
    unsigned short* wihD = (unsigned short*)(ws + o); o += nLW * 2;
    unsigned short* whhD = (unsigned short*)(ws + o); o += nLW * 2;
    unsigned short* fcWb = (unsigned short*)(ws + o); o += (size_t)VV * HH * 2;
    unsigned short* X0   = (unsigned short*)(ws + o); o += (size_t)TT * BB * HH * 2;
    unsigned short* X1   = (unsigned short*)(ws + o); o += (size_t)TT * BB * HH * 2;
    float* precomp       = (float*)(ws + o);          o += (size_t)TT * BB * G4H * 4;
    float* h_all         = (float*)(ws + o);          o += (size_t)LL * BB * HH * 4;
    float* c_all         = (float*)(ws + o);          o += (size_t)LL * BB * HH * 4;

    // ---- 1) weight conversions to bf16 ----
    {
        int n = (int)nLW;
        int g = (n + 255) / 256;
        k_f32_to_bf16<<<g, 256, 0, stream>>>(enc_Wih, wihE, n);
        k_f32_to_bf16<<<g, 256, 0, stream>>>(enc_Whh, whhE, n);
        k_f32_to_bf16<<<g, 256, 0, stream>>>(dec_Wih, wihD, n);
        k_f32_to_bf16<<<g, 256, 0, stream>>>(dec_Whh, whhD, n);
        int nf = VV * HH;
        k_f32_to_bf16<<<(nf + 255) / 256, 256, 0, stream>>>(fc_W, fcWb, nf);
    }

    // ---- 2) initial h/c (broadcast over batch) ----
    k_init_state<<<(LL * BB * HH + 255) / 256, 256, 0, stream>>>(hidden0, cell0, h_all, c_all);

    // ---- 3) encoder input embedding ----
    k_embed<<<(TT * BB * HH + 255) / 256, 256, 0, stream>>>(inputs, enc_emb, X0, 0);

    unsigned short* Xin  = X0;
    unsigned short* Xout = X1;

    // 256x64 C patch per block (8 waves x 32x64); all grids exact.
    const int    gridPre = (TT * BB / 256) * (G4H / 64);        // 16*16  = 256 blocks
    const size_t ldsRec  = (size_t)BB * HH * 4 + 2 * (size_t)BB * HH * 2; // 128KB

    for (int phase = 0; phase < 2; ++phase) {
        unsigned short* Wih = phase ? wihD : wihE;
        unsigned short* Whh = phase ? whhD : whhE;
        const float* bih = phase ? dec_bih : enc_bih;
        const float* bhh = phase ? dec_bhh : enc_bhh;

        if (phase == 1)  // decoder input: targets shifted right, zero start token
            k_embed<<<(TT * BB * HH + 255) / 256, 256, 0, stream>>>(targets, dec_emb, Xin, 1);

        for (int l = 0; l < LL; ++l) {
            // batched input projection for all T: precomp = Xin @ Wih[l]^T + bih + bhh
            k_gemm_blk<<<gridPre, 256, 0, stream>>>(
                Xin, Wih + (size_t)l * G4H * HH, precomp,
                TT * BB, G4H, bih + l * G4H, bhh + l * G4H);

            // full-layer recurrence in one persistent workgroup
            k_lstm_layer<<<1, 1024, (unsigned)ldsRec, stream>>>(
                Whh + (size_t)l * G4H * HH, precomp,
                h_all + (size_t)l * BB * HH, c_all + (size_t)l * BB * HH, Xout);

            unsigned short* tmp = Xin; Xin = Xout; Xout = tmp;
        }
        // encoder final h_all/c_all directly seed the decoder (no copy needed)
    }

    // ---- 4) final FC straight into d_out: logits = Ydec @ fc_W^T + fc_b ----
    float* logits = (float*)d_out;
    const int gridFC = (TT * BB / 256) * (VV / 64);             // 16*500 = 8000 blocks
    k_gemm_blk<<<gridFC, 256, 0, stream>>>(
        Xin, fcWb, logits, TT * BB, VV, fc_b, nullptr);

    // ---- 5) argmax -> predicted_words appended after logits ----
    k_argmax<<<(TT * BB + 7) / 8, 256, 0, stream>>>(logits, logits + (size_t)TT * BB * VV);
}
